// AttnDecoderRNN_28003186770477
// MI455X (gfx1250) — compile-verified, matrix operand order checked by reading, not run on hardware
//
#include <hip/hip_runtime.h>

#define BB 32
#define SS 64
#define TT 64
#define HH 512
#define VV 32000

typedef float v8f __attribute__((ext_vector_type(8)));
typedef __bf16 v16bf __attribute__((ext_vector_type(16)));

union FragU {
    unsigned int u[8];
    v16bf v;
};

__device__ __forceinline__ unsigned int f2bf_pack(float a, float b) {
    unsigned int ua = __float_as_uint(a);
    unsigned int ub = __float_as_uint(b);
    unsigned int ra = (ua + 0x7FFFu + ((ua >> 16) & 1u)) >> 16;   // RNE f32->bf16
    unsigned int rb = (ub + 0x7FFFu + ((ub >> 16) & 1u)) >> 16;
    return (ra & 0xFFFFu) | (rb << 16);
}

// Pack-convert f32 pairs -> bf16x2 (one-time per call for weights/activations).
__global__ __launch_bounds__(256)
void cvt_f32_bf16_kernel(const float* __restrict__ src, unsigned int* __restrict__ dst,
                         int npairs) {
    int i = blockIdx.x * 256 + threadIdx.x;
    if (i < npairs) {
        const float2 f = *(const float2*)&src[2 * i];
        dst[i] = f2bf_pack(f.x, f.y);
    }
}

// ---------------------------------------------------------------------------
// C[M,N] = A[M,K] @ W[N,K]^T + bias[N]; A/W are bf16, C is f32.
// Block tile 32(M) x 256(N), BK=32, 256 threads = 8 waves. Wave w owns N cols
// [32w,32w+32): 2 N-tiles x 2 M-tiles = 4 v_wmma_f32_16x16x32_bf16 per K step.
// Tiles staged memory->LDS with GLOBAL_LOAD_ASYNC_TO_LDS_B128 (ASYNCcnt).
// Requires M%32==0, N%256==0, K%32==0 (true for all shapes used here).
// Optional output row remap: orow = (row % map_mod)*map_mul + row/map_mod.
// ---------------------------------------------------------------------------
#define LDT 20   // uints per LDS tile row: 16 data (32 bf16) + 4 pad; 80B keeps 16B align

__global__ __launch_bounds__(256)
void wmma_gemm_bf16(const unsigned short* __restrict__ A, const unsigned short* __restrict__ W,
                    const float* __restrict__ bias, float* __restrict__ out,
                    int M, int N, int K, int map_mod, int map_mul) {
    __shared__ unsigned int sA[32 * LDT];
    __shared__ unsigned int sB[256 * LDT];

    const int tid  = threadIdx.x;
    const int wave = tid >> 5;
    const int lane = tid & 31;
    const int n0 = blockIdx.x * 256;
    const int m0 = blockIdx.y * 32;

    v8f acc00 = {0.f,0.f,0.f,0.f,0.f,0.f,0.f,0.f};
    v8f acc01 = acc00, acc10 = acc00, acc11 = acc00;

    const int rl = lane & 15;
    const int g0 = (lane >> 4) ? 4 : 0;    // uint offsets selecting K-halves
    const int g1 = (lane >> 4) ? 12 : 8;

    const char* Ab = (const char*)A;
    const char* Wb = (const char*)W;

    for (int k0 = 0; k0 < K; k0 += 32) {
        // --- async stage W tile: 256 rows x 64B = 1024 x 16B transfers ---
        #pragma unroll
        for (int i = 0; i < 4; ++i) {
            int p = tid + i * 256;
            int row = p >> 2, seg = p & 3;
            const void* g = Wb + ((size_t)(n0 + row) * K + k0 + seg * 8) * 2;
            unsigned ldst = (unsigned)(uintptr_t)&sB[row * LDT + seg * 4];
            asm volatile("global_load_async_to_lds_b128 %0, %1, off"
                         :: "v"(ldst), "v"(g) : "memory");
        }
        // --- async stage A tile: 32 rows x 64B = 128 x 16B transfers ---
        if (tid < 128) {
            int row = tid >> 2, seg = tid & 3;
            const void* g = Ab + ((size_t)(m0 + row) * K + k0 + seg * 8) * 2;
            unsigned ldst = (unsigned)(uintptr_t)&sA[row * LDT + seg * 4];
            asm volatile("global_load_async_to_lds_b128 %0, %1, off"
                         :: "v"(ldst), "v"(g) : "memory");
        }
        asm volatile("s_wait_asynccnt 0x0" ::: "memory");
        __syncthreads();

        // B fragments: cols n = n0 + 32*wave + {0..15, 16..31}; B[k][n] = W[n][k]
        FragU fb0, fb1, fa0, fa1;
        {
            const unsigned int* rowp = &sB[(wave * 32 + rl) * LDT];
            *(uint4*)&fb0.u[0] = *(const uint4*)&rowp[g0];
            *(uint4*)&fb0.u[4] = *(const uint4*)&rowp[g1];
            rowp = &sB[(wave * 32 + 16 + rl) * LDT];
            *(uint4*)&fb1.u[0] = *(const uint4*)&rowp[g0];
            *(uint4*)&fb1.u[4] = *(const uint4*)&rowp[g1];
            rowp = &sA[rl * LDT];
            *(uint4*)&fa0.u[0] = *(const uint4*)&rowp[g0];
            *(uint4*)&fa0.u[4] = *(const uint4*)&rowp[g1];
            rowp = &sA[(16 + rl) * LDT];
            *(uint4*)&fa1.u[0] = *(const uint4*)&rowp[g0];
            *(uint4*)&fa1.u[4] = *(const uint4*)&rowp[g1];
        }

        acc00 = __builtin_amdgcn_wmma_f32_16x16x32_bf16(false, fa0.v, false, fb0.v,
                                                        (short)0, acc00, false, false);
        acc01 = __builtin_amdgcn_wmma_f32_16x16x32_bf16(false, fa0.v, false, fb1.v,
                                                        (short)0, acc01, false, false);
        acc10 = __builtin_amdgcn_wmma_f32_16x16x32_bf16(false, fa1.v, false, fb0.v,
                                                        (short)0, acc10, false, false);
        acc11 = __builtin_amdgcn_wmma_f32_16x16x32_bf16(false, fa1.v, false, fb1.v,
                                                        (short)0, acc11, false, false);
        __syncthreads();
    }

    // Write back per 16x16 f32 C layout: lane -> col (lane&15), rows r (+8 hi half)
    const int rlo = (lane >> 4) * 8;
    #pragma unroll
    for (int nt = 0; nt < 2; ++nt) {
        const int col = n0 + wave * 32 + nt * 16 + rl;
        const float bv = bias[col];
        const v8f aL = nt ? acc01 : acc00;
        const v8f aH = nt ? acc11 : acc10;
        #pragma unroll
        for (int r = 0; r < 8; ++r) {
            int grow = m0 + rlo + r;
            int orow = map_mod ? ((grow % map_mod) * map_mul + grow / map_mod) : grow;
            out[(size_t)orow * N + col] = aL[r] + bv;
        }
        #pragma unroll
        for (int r = 0; r < 8; ++r) {
            int grow = m0 + 16 + rlo + r;
            int orow = map_mod ? ((grow % map_mod) * map_mul + grow / map_mod) : grow;
            out[(size_t)orow * N + col] = aH[r] + bv;
        }
    }
}

// ---------------------------------------------------------------------------
// Per-step attention: scores = Va . tanh(q + Ua_keys), softmax over S,
// ctx = sum_s w_s * keys; assemble x = [emb[token], ctx] as packed bf16.
// One block per batch element.
// ---------------------------------------------------------------------------
__global__ __launch_bounds__(256)
void attn_ctx_kernel(const float* __restrict__ q, const float* __restrict__ Ua_keys,
                     const float* __restrict__ Va_w, const float* __restrict__ Va_b,
                     const float* __restrict__ keys, const float* __restrict__ emb,
                     const int* __restrict__ target, unsigned int* __restrict__ x_bf,
                     float* __restrict__ attn_out, int t) {
    __shared__ float sQ[HH];
    __shared__ float sV[HH];
    __shared__ float sW[SS];
    const int tid = threadIdx.x;
    const int b = blockIdx.x;
    const int wave = tid >> 5, lane = tid & 31;

    for (int h = tid; h < HH; h += 256) {
        sQ[h] = q[b * HH + h];
        sV[h] = Va_w[h];
    }
    __syncthreads();

    for (int si = 0; si < 8; ++si) {
        int s = wave * 8 + si;
        const float* uk = &Ua_keys[((size_t)b * SS + s) * HH];
        float p = 0.f;
        for (int h = lane; h < HH; h += 32) p += sV[h] * tanhf(sQ[h] + uk[h]);
        for (int off = 16; off; off >>= 1) p += __shfl_down(p, off, 32);
        if (lane == 0) sW[s] = p + Va_b[0];
    }
    __syncthreads();

    if (tid == 0) {
        float m = -3.4e38f;
        for (int s = 0; s < SS; ++s) m = fmaxf(m, sW[s]);
        float sum = 0.f;
        for (int s = 0; s < SS; ++s) { float e = __expf(sW[s] - m); sW[s] = e; sum += e; }
        float inv = 1.f / sum;
        for (int s = 0; s < SS; ++s) sW[s] *= inv;
    }
    __syncthreads();

    if (tid < SS) attn_out[((size_t)b * TT + t) * SS + tid] = sW[tid];

    const int tok = (t == 0) ? 0 : target[b * TT + t - 1];
    // Each thread produces one bf16x2 pair of emb and of ctx (h = 2*tid, 2*tid+1).
    {
        const int h = 2 * tid;
        const float e0 = emb[(size_t)tok * HH + h];
        const float e1 = emb[(size_t)tok * HH + h + 1];
        x_bf[b * (HH) + tid] = f2bf_pack(e0, e1);            // x row = 1024 bf16 = 512 uints

        float c0 = 0.f, c1 = 0.f;
        const float* kb = &keys[(size_t)b * SS * HH + h];
        #pragma unroll 4
        for (int s = 0; s < SS; ++s) {
            float w = sW[s];
            c0 += w * kb[(size_t)s * HH];
            c1 += w * kb[(size_t)s * HH + 1];
        }
        x_bf[b * (HH) + 256 + tid] = f2bf_pack(c0, c1);
    }
}

// ---------------------------------------------------------------------------
// GRU cell elementwise (PyTorch convention). Writes f32 h_next plus packed
// bf16 h_next (for next step's WMMA GEMMs) and bf16 H_all[t] (row = t*B+b).
// ---------------------------------------------------------------------------
__global__ __launch_bounds__(256)
void gru_cell_kernel(const float* __restrict__ gi, const float* __restrict__ gh,
                     const float* __restrict__ h_cur, float* __restrict__ h_next,
                     unsigned int* __restrict__ h_next_bf, unsigned int* __restrict__ Hall_bf,
                     int t) {
    int idx = blockIdx.x * 256 + threadIdx.x;   // 32*256 pairs total
    int b = idx >> 8;
    int jp = idx & 255;
    int j = jp * 2;
    float hv[2];
    #pragma unroll
    for (int e = 0; e < 2; ++e) {
        int jj = j + e;
        float ir = gi[b * 3 * HH + jj];
        float iz = gi[b * 3 * HH + HH + jj];
        float in = gi[b * 3 * HH + 2 * HH + jj];
        float hr = gh[b * 3 * HH + jj];
        float hz = gh[b * 3 * HH + HH + jj];
        float hn = gh[b * 3 * HH + 2 * HH + jj];
        float r = 1.f / (1.f + __expf(-(ir + hr)));
        float z = 1.f / (1.f + __expf(-(iz + hz)));
        float n = tanhf(in + r * hn);
        hv[e] = (1.f - z) * n + z * h_cur[b * HH + jj];
    }
    *(float2*)&h_next[b * HH + j] = make_float2(hv[0], hv[1]);
    unsigned p = f2bf_pack(hv[0], hv[1]);
    h_next_bf[b * 256 + jp] = p;
    Hall_bf[((size_t)t * BB + b) * 256 + jp] = p;
}

// ---------------------------------------------------------------------------
// In-place log_softmax over V per row of d_out's [B*T, V] logits region.
// ---------------------------------------------------------------------------
__global__ __launch_bounds__(256)
void log_softmax_kernel(float* __restrict__ out) {
    __shared__ float red[256];
    const int tid = threadIdx.x;
    float* row = out + (size_t)blockIdx.x * VV;

    float m = -3.4e38f;
    for (int i = tid; i < VV; i += 256) m = fmaxf(m, row[i]);
    red[tid] = m; __syncthreads();
    for (int s = 128; s; s >>= 1) { if (tid < s) red[tid] = fmaxf(red[tid], red[tid + s]); __syncthreads(); }
    m = red[0]; __syncthreads();

    float sum = 0.f;
    for (int i = tid; i < VV; i += 256) sum += __expf(row[i] - m);
    red[tid] = sum; __syncthreads();
    for (int s = 128; s; s >>= 1) { if (tid < s) red[tid] += red[tid + s]; __syncthreads(); }
    const float lse = m + __logf(red[0]);

    for (int i = tid; i < VV; i += 256) row[i] -= lse;
}

// ---------------------------------------------------------------------------
extern "C" void kernel_launch(void* const* d_in, const int* in_sizes, int n_in,
                              void* d_out, int out_size, void* d_ws, size_t ws_size,
                              hipStream_t stream) {
    const float* keys   = (const float*)d_in[0];   // [B,S,H]
    const float* h0     = (const float*)d_in[1];   // [1,B,H]
    const int*   target = (const int*)d_in[2];     // [B,T]
    // d_in[3] = max_len (compile-time TT)
    const float* emb    = (const float*)d_in[4];   // [V,H]
    const float* Wa_w   = (const float*)d_in[5];
    const float* Wa_b   = (const float*)d_in[6];
    const float* Ua_w   = (const float*)d_in[7];
    const float* Ua_b   = (const float*)d_in[8];
    const float* Va_w   = (const float*)d_in[9];
    const float* Va_b   = (const float*)d_in[10];
    const float* W_ih   = (const float*)d_in[11];  // [3H,2H]
    const float* W_hh   = (const float*)d_in[12];  // [3H,H]
    const float* b_ih   = (const float*)d_in[13];
    const float* b_hh   = (const float*)d_in[14];
    const float* out_w  = (const float*)d_in[15];  // [V,H]
    const float* out_b  = (const float*)d_in[16];

    // -------- workspace layout --------
    unsigned int* wsu = (unsigned int*)d_ws;
    unsigned int* outw_bf = wsu;                       // VV*HH/2      = 8,192,000
    unsigned int* wih_bf  = outw_bf + (size_t)VV * HH / 2;   // 786,432
    unsigned int* whh_bf  = wih_bf + 3 * HH * 2 * HH / 2;    // 393,216
    unsigned int* wa_bf   = whh_bf + 3 * HH * HH / 2;        // 131,072
    unsigned int* ua_bf   = wa_bf + HH * HH / 2;             // 131,072
    unsigned int* keys_bf = ua_bf + HH * HH / 2;             // 524,288
    unsigned int* hall_bf = keys_bf + BB * SS * HH / 2;      // 524,288
    unsigned int* hbf0    = hall_bf + (size_t)BB * TT * HH / 2;  // 8,192
    unsigned int* hbf1    = hbf0 + BB * HH / 2;
    unsigned int* xbf     = hbf1 + BB * HH / 2;              // 16,384
    float* wsf   = (float*)(xbf + BB * 2 * HH / 2);
    float* UaK   = wsf;                     // 2048*512
    float* hf0   = UaK + (size_t)BB * SS * HH;
    float* hf1   = hf0 + BB * HH;
    float* qbuf  = hf1 + BB * HH;
    float* gibuf = qbuf + BB * HH;
    float* ghbuf = gibuf + BB * 3 * HH;

    float* out_logits = (float*)d_out;                         // [B,T,V]
    float* out_hT     = (float*)d_out + (size_t)BB * TT * VV;  // [1,B,H]
    float* out_attn   = out_hT + BB * HH;                      // [B,T,S]

    dim3 blk(256);
    // -------- one-time (per call) bf16 conversions --------
    cvt_f32_bf16_kernel<<<VV * HH / 512, blk, 0, stream>>>(out_w, outw_bf, VV * HH / 2);
    cvt_f32_bf16_kernel<<<3 * HH * 2 * HH / 512, blk, 0, stream>>>(W_ih, wih_bf, 3 * HH * 2 * HH / 2);
    cvt_f32_bf16_kernel<<<3 * HH * HH / 512, blk, 0, stream>>>(W_hh, whh_bf, 3 * HH * HH / 2);
    cvt_f32_bf16_kernel<<<HH * HH / 512, blk, 0, stream>>>(Wa_w, wa_bf, HH * HH / 2);
    cvt_f32_bf16_kernel<<<HH * HH / 512, blk, 0, stream>>>(Ua_w, ua_bf, HH * HH / 2);
    cvt_f32_bf16_kernel<<<BB * SS * HH / 512, blk, 0, stream>>>(keys, keys_bf, BB * SS * HH / 2);
    cvt_f32_bf16_kernel<<<BB * HH / 512, blk, 0, stream>>>(h0, hbf0, BB * HH / 2);
    hipMemcpyAsync(hf0, h0, (size_t)BB * HH * sizeof(float), hipMemcpyDeviceToDevice, stream);

    // Ua_keys = keys @ Ua_w.T + Ua_b : [2048,512] x [512,512]
    wmma_gemm_bf16<<<dim3(HH / 256, (BB * SS) / 32), blk, 0, stream>>>(
        (const unsigned short*)keys_bf, (const unsigned short*)ua_bf, Ua_b, UaK,
        BB * SS, HH, HH, 0, 0);

    float* hcf = hf0; float* hnf = hf1;
    unsigned int* hcb = hbf0; unsigned int* hnb = hbf1;
    for (int t = 0; t < TT; ++t) {
        // q = h @ Wa.T + Wa_b
        wmma_gemm_bf16<<<dim3(HH / 256, 1), blk, 0, stream>>>(
            (const unsigned short*)hcb, (const unsigned short*)wa_bf, Wa_b, qbuf,
            BB, HH, HH, 0, 0);
        // attention + x assembly (bf16 packed)
        attn_ctx_kernel<<<BB, blk, 0, stream>>>(
            qbuf, UaK, Va_w, Va_b, keys, emb, target, xbf, out_attn, t);
        // gi = x @ W_ih.T + b_ih
        wmma_gemm_bf16<<<dim3((3 * HH) / 256, 1), blk, 0, stream>>>(
            (const unsigned short*)xbf, (const unsigned short*)wih_bf, b_ih, gibuf,
            BB, 3 * HH, 2 * HH, 0, 0);
        // gh = h @ W_hh.T + b_hh
        wmma_gemm_bf16<<<dim3((3 * HH) / 256, 1), blk, 0, stream>>>(
            (const unsigned short*)hcb, (const unsigned short*)whh_bf, b_hh, ghbuf,
            BB, 3 * HH, HH, 0, 0);
        // GRU cell -> h_next (f32 + bf16), H_all[t] (bf16)
        gru_cell_kernel<<<(BB * HH / 2) / 256, blk, 0, stream>>>(
            gibuf, ghbuf, hcf, hnf, hnb, hall_bf, t);
        { float* tf = hcf; hcf = hnf; hnf = tf; }
        { unsigned int* tb = hcb; hcb = hnb; hnb = tb; }
    }
    hipMemcpyAsync(out_hT, hcf, (size_t)BB * HH * sizeof(float),
                   hipMemcpyDeviceToDevice, stream);

    // Batched logits = H_all @ out_w.T + out_b, remap row t*B+b -> b*T+t
    wmma_gemm_bf16<<<dim3(VV / 256, (BB * TT) / 32), blk, 0, stream>>>(
        (const unsigned short*)hall_bf, (const unsigned short*)outw_bf, out_b, out_logits,
        BB * TT, VV, HH, BB, TT);

    // In-place log_softmax over V
    log_softmax_kernel<<<BB * TT, blk, 0, stream>>>(out_logits);
}
